// GAE_39256001085529
// MI455X (gfx1250) — matrix operand order repaired and checked by reading, not compiled, not run on hardware
//
#include <hip/hip_runtime.h>
#include <hip/hip_bf16.h>
#include <hip/hip_fp16.h>

typedef __attribute__((ext_vector_type(16))) _Float16 v16h;
typedef __attribute__((ext_vector_type(8)))  _Float16 v8h;
typedef __attribute__((ext_vector_type(8)))  float    v8f;

#define NNODES 16384
#define NEDGES 524288

// ---------------------------------------------------------------- gather ----
__global__ void gather_emb_kernel(const int* __restrict__ ids,
                                  const float* __restrict__ emb,
                                  float* __restrict__ h0) {
  int t = blockIdx.x * blockDim.x + threadIdx.x;      // exactly N*128 threads
  int node = t >> 7;
  int f    = t & 127;
  h0[t] = emb[((size_t)ids[node] << 7) + f];
}

// ---------------------------------------------------------------- degree ----
__global__ void degree_kernel(const int* __restrict__ dst,
                              float* __restrict__ deg, int E) {
  int e = blockIdx.x * blockDim.x + threadIdx.x;
  if (e < E) atomicAdd(&deg[dst[e]], 1.0f);
}

// ------------------------------------------------- edge message scatter -----
// one thread per (edge, 4-feature group); float4 read, 4x global_atomic_add_f32
template<int D>
__global__ void scatter_msgs_kernel(const float* __restrict__ h,
                                    const int* __restrict__ src,
                                    const int* __restrict__ dst,
                                    const float* __restrict__ w,
                                    float* __restrict__ neigh) {
  constexpr int G = D / 4;
  int t  = blockIdx.x * blockDim.x + threadIdx.x;     // exactly E*G threads
  int e  = t / G;
  int fg = (t % G) << 2;
  int s = src[e], d = dst[e];
  float we = w[e];
  const float4 hv = *reinterpret_cast<const float4*>(h + (size_t)s * D + fg);
  float* o = neigh + (size_t)d * D + fg;
  atomicAdd(o + 0, hv.x * we);
  atomicAdd(o + 1, hv.y * we);
  atomicAdd(o + 2, hv.z * we);
  atomicAdd(o + 3, hv.w * we);
}

// ------------------------------- fused SAGE GEMM + bias + LayerNorm --------
// one block per node; blockDim == DOUT; weight reads coalesced along j
template<int DIN, int DOUT>
__global__ void sage_ln_kernel(const float* __restrict__ h,
                               const float* __restrict__ neigh,
                               const float* __restrict__ deg,
                               const float* __restrict__ Ws,
                               const float* __restrict__ Wn,
                               const float* __restrict__ bias,
                               const float* __restrict__ g,
                               const float* __restrict__ b,
                               float* __restrict__ out) {
  __shared__ float hrow[DIN];
  __shared__ float nrow[DIN];
  __shared__ float rsum[DOUT];
  __shared__ float rsq[DOUT];
  int node = blockIdx.x;
  int j    = threadIdx.x;
  float dinv = 1.0f / fmaxf(deg[node], 1.0f);
  for (int k = j; k < DIN; k += DOUT) {
    hrow[k] = h[(size_t)node * DIN + k];
    nrow[k] = neigh[(size_t)node * DIN + k] * dinv;
  }
  __syncthreads();
  float acc = bias[j];
#pragma unroll 8
  for (int k = 0; k < DIN; ++k)
    acc = fmaf(hrow[k], Ws[k * DOUT + j], fmaf(nrow[k], Wn[k * DOUT + j], acc));
  rsum[j] = acc;
  rsq[j]  = acc * acc;
  __syncthreads();
  for (int s = DOUT >> 1; s > 0; s >>= 1) {
    if (j < s) { rsum[j] += rsum[j + s]; rsq[j] += rsq[j + s]; }
    __syncthreads();
  }
  float mean = rsum[0] * (1.0f / DOUT);
  float var  = rsq[0] * (1.0f / DOUT) - mean * mean;
  out[(size_t)node * DOUT + j] = (acc - mean) * rsqrtf(var + 1e-5f) * g[j] + b[j];
}

// ---------------------------------------------------------------- f32→f16 --
__global__ void f32_to_f16_kernel(const float* __restrict__ in,
                                  _Float16* __restrict__ out) {
  int t = blockIdx.x * blockDim.x + threadIdx.x;      // exactly N*64 threads
  out[t] = (_Float16)in[t];
}

// ------------------------------- decoder: adj = sigmoid(H @ H^T), WMMA -----
// H: [N,64] f16 row-major (2 MB, L2-resident). 16x16 tiles, K=64 -> two
// v_wmma_f32_16x16x32_f16 per tile. 8 waves/block; wave owns one M tile and
// sweeps 4 N tiles, reusing its A fragments. Sigmoid = v_exp + v_add + v_rcp
// (hardware reciprocal; avoids IEEE-div Newton-Raphson chain).
__global__ void __launch_bounds__(256)
adj_sigmoid_wmma_kernel(const _Float16* __restrict__ H, float* __restrict__ adj) {
  const int lane   = threadIdx.x & 31;
  const int wave   = threadIdx.x >> 5;
  const int mtile  = blockIdx.y * 8 + wave;           // 0..1023
  const int ntile0 = blockIdx.x * 4;

  // A fragment (16x32 f16): lanes 0-15 -> M=lane, K in {0..7,16..23};
  // lanes 16-31 -> M=lane-16, K in {8..15,24..31}. Two 16B loads per chunk.
  const int mrow  = (mtile << 4) + (lane & 15);
  const int kbase = (lane >> 4) << 3;                 // 0 or 8
  const int klo   = (lane >> 4) << 4;                 // 0 or 16 (B frag)
  const _Float16* rowA = H + (size_t)mrow * 64;

  union V16 { v16h v; v8h h[2]; };
  V16 a0, a1;
  a0.h[0] = *reinterpret_cast<const v8h*>(rowA + kbase);
  a0.h[1] = *reinterpret_cast<const v8h*>(rowA + 16 + kbase);
  a1.h[0] = *reinterpret_cast<const v8h*>(rowA + 32 + kbase);
  a1.h[1] = *reinterpret_cast<const v8h*>(rowA + 48 + kbase);

  const int mlo = (lane >> 4) << 3;                   // D-frag row offset: 0 or 8

  for (int t = 0; t < 4; ++t) {
    const int ntile = ntile0 + t;
    const int ncol  = (ntile << 4) + (lane & 15);
    // B fragment (32x16 f16): lane holds column n; K contiguous in memory
    // since B[k][n] = H[n][k]. K order matches A: {kbase..+7, 16+kbase..+7}.
    const _Float16* rowB = H + (size_t)ncol * 64;
    V16 b0, b1;
    b0.h[0] = *reinterpret_cast<const v8h*>(rowB + kbase);
    b0.h[1] = *reinterpret_cast<const v8h*>(rowB + 16 + kbase);
    b1.h[0] = *reinterpret_cast<const v8h*>(rowB + 32 + kbase);
    b1.h[1] = *reinterpret_cast<const v8h*>(rowB + 48 + kbase);

    v8f c = {};
    c = __builtin_amdgcn_wmma_f32_16x16x32_f16(false, a0.v, false, b0.v,
                                               (short)0, c, false, false);
    c = __builtin_amdgcn_wmma_f32_16x16x32_f16(false, a1.v, false, b1.v,
                                               (short)0, c, false, false);

    // D layout: VGPR r -> row (mlo + r), column = lane%16
    float* outbase = adj + (size_t)((mtile << 4) + mlo) * NNODES + ncol;
#pragma unroll
    for (int r = 0; r < 8; ++r) {
      float v = c[r];
      float s = __builtin_amdgcn_rcpf(1.0f + __expf(-v));   // v_rcp_f32
      outbase[(size_t)r * NNODES] = s;
    }
  }
}

// ---------------------------------------------------------------------------
extern "C" void kernel_launch(void* const* d_in, const int* in_sizes, int n_in,
                              void* d_out, int out_size, void* d_ws, size_t ws_size,
                              hipStream_t stream) {
  (void)in_sizes; (void)n_in; (void)out_size; (void)ws_size;

  const int*   node_ids = (const int*)  d_in[0];
  const int*   src      = (const int*)  d_in[1];
  const int*   dst      = (const int*)  d_in[2];
  const float* ew       = (const float*)d_in[3];
  const float* emb      = (const float*)d_in[4];
  const float* Ws0 = (const float*)d_in[5];
  const float* Wn0 = (const float*)d_in[6];
  const float* bi0 = (const float*)d_in[7];
  const float* g0  = (const float*)d_in[8];
  const float* lb0 = (const float*)d_in[9];
  const float* Ws1 = (const float*)d_in[10];
  const float* Wn1 = (const float*)d_in[11];
  const float* bi1 = (const float*)d_in[12];
  const float* g1  = (const float*)d_in[13];
  const float* lb1 = (const float*)d_in[14];

  char* ws = (char*)d_ws;
  float*    h0  = (float*)(ws);                                    // 8 MiB
  float*    h1  = (float*)(ws + ((size_t)8  << 20));               // 8 MiB
  float*    ng0 = (float*)(ws + ((size_t)16 << 20));               // 8 MiB
  float*    ng1 = (float*)(ws + ((size_t)24 << 20));               // 8 MiB
  float*    deg = (float*)(ws + ((size_t)32 << 20));               // 64 KiB
  _Float16* hf  = (_Float16*)(ws + ((size_t)32 << 20) + (64 << 10)); // 2 MiB

  float* h_out = (float*)d_out;                    // [N,64] f32
  float* adj   = h_out + (size_t)NNODES * 64;      // [N,N]  f32

  hipMemsetAsync(ng0, 0, (size_t)NNODES * 128 * sizeof(float), stream);
  hipMemsetAsync(ng1, 0, (size_t)NNODES * 128 * sizeof(float), stream);
  hipMemsetAsync(deg, 0, (size_t)NNODES * sizeof(float), stream);

  gather_emb_kernel<<<(NNODES * 128) / 256, 256, 0, stream>>>(node_ids, emb, h0);
  degree_kernel<<<NEDGES / 256, 256, 0, stream>>>(dst, deg, NEDGES);

  scatter_msgs_kernel<128><<<(NEDGES * 32) / 256, 256, 0, stream>>>(h0, src, dst, ew, ng0);
  sage_ln_kernel<128, 128><<<NNODES, 128, 0, stream>>>(h0, ng0, deg, Ws0, Wn0,
                                                       bi0, g0, lb0, h1);

  scatter_msgs_kernel<128><<<(NEDGES * 32) / 256, 256, 0, stream>>>(h1, src, dst, ew, ng1);
  sage_ln_kernel<128, 64><<<NNODES, 64, 0, stream>>>(h1, ng1, deg, Ws1, Wn1,
                                                     bi1, g1, lb1, h_out);

  f32_to_f16_kernel<<<(NNODES * 64) / 256, 256, 0, stream>>>(h_out, hf);

  dim3 grid(1024 / 4, 1024 / 8);                   // (N/16)/TN x (N/16)/waves
  adj_sigmoid_wmma_kernel<<<grid, 256, 0, stream>>>(hf, adj);
}
